// VLTF_42640435315316
// MI455X (gfx1250) — matrix operand order repaired and checked by reading, not compiled, not run on hardware
//
#include <hip/hip_runtime.h>

// =====================================================================
// VLTF fuse block for MI455X (gfx1250, wave32, WMMA).
//
// d_in flattening (setup_inputs dict order, params flattened in insertion
// order):
//  0 src (16,4096,512)  1 lan (16,20,768)  2 pos_embed (UNUSED)  3 lan_attmask (16,20)
//  4 mem_tok  5 neg_mem_tok  6 in_w  7 in_b  8 in_bn_g  9 in_bn_b
// 10 lan_w 11 lan_b 12 out_w 13 out_b 14 out_bn_g 15 out_bn_b
// 16..23 vl_{q,k,v,o}{w,b}   24..31 mf_{q,k,v,o}{w,b}   32..39 ff_{q,k,v,o}{w,b}
// (vl_vw/vl_ow unused: vlf is computed but never returned by the reference.)
//
// d_out layout (floats, concatenated in return order):
//   out      @ 0          (16*4096*512 = 33554432)
//   lan_mem  @ 33554432   (16*2*512    = 16384)
//   vl_att   @ 33570816   (16*8*20*4096= 10485760)
//   mem_att  @ 44056576   (16*8*20     = 2560)
//   f_att    @ 44059136   (16*8*4096*2 = 1048576)
// =====================================================================

typedef _Float16 half_t;
typedef __attribute__((ext_vector_type(16))) _Float16 v16h;
typedef __attribute__((ext_vector_type(8)))  float    v8f;
typedef __attribute__((ext_vector_type(4)))  unsigned int u32x4;

union H16 { v16h h; u32x4 u[2]; };

#define EPSN 1e-5f

// ---------------------------------------------------------------------
// f32 -> f16 cast
__global__ __launch_bounds__(256)
void vltf_cvt_kernel(const float* __restrict__ in, half_t* __restrict__ out, size_t n) {
  size_t i = (size_t)blockIdx.x * blockDim.x + threadIdx.x;
  if (i < n) out[i] = (half_t)in[i];
}

// ---------------------------------------------------------------------
// fold eval-BatchNorm into GEMM epilogue: alpha = g/sqrt(1+eps),
// beta = b_lin*alpha + b_bn   (for both input and output BN)
__global__ __launch_bounds__(256)
void vltf_prep_kernel(const float* g1, const float* bl1, const float* bn1,
                      const float* g2, const float* bl2, const float* bn2,
                      float* a1, float* be1, float* a2, float* be2) {
  int n = blockIdx.x * blockDim.x + threadIdx.x;
  if (n >= 512) return;
  float rs = rsqrtf(1.0f + EPSN);
  float A1 = g1[n] * rs; a1[n] = A1; be1[n] = bl1[n] * A1 + bn1[n];
  float A2 = g2[n] * rs; a2[n] = A2; be2[n] = bl2[n] * A2 + bn2[n];
}

#define GLDT 40   // LDS row stride in halves (80B; keeps b128 alignment)

// ---------------------------------------------------------------------
// Small-M WMMA GEMM: BM=BN=64, BK=32, 128 threads = 4 waves; each wave a
// 16x64 strip. Used for the 320-row language-path GEMMs.
__global__ __launch_bounds__(128)
void vltf_gemm_wmma64(const half_t* __restrict__ A, const half_t* __restrict__ Wt,
                      const float* __restrict__ bias, const float* __restrict__ scale,
                      float* __restrict__ outF, half_t* __restrict__ outH,
                      int M, int N, int K) {
  __shared__ alignas(16) half_t As[64 * GLDT];
  __shared__ alignas(16) half_t Bs[64 * GLDT];

  const int tid  = threadIdx.x;
  const int wave = tid >> 5;
  const int lane = tid & 31;
  const int mBase = blockIdx.x * 64;
  const int nBase = blockIdx.y * 64;

  v8f acc[4];
#pragma unroll
  for (int i = 0; i < 4; ++i)
#pragma unroll
    for (int j = 0; j < 8; ++j) acc[i][j] = 0.0f;

  const int lrow = tid >> 1;          // 0..63
  const int lseg = (tid & 1) * 16;    // 0 or 16 (halves)
  const int fr  = lane & 15;
  const int fhi = lane >> 4;

  for (int k0 = 0; k0 < K; k0 += 32) {
    {
      const half_t* ga = A  + (size_t)(mBase + lrow) * K + k0 + lseg;
      const half_t* gb = Wt + (size_t)(nBase + lrow) * K + k0 + lseg;
      u32x4 a0 = *(const u32x4*)ga;
      u32x4 a1 = *(const u32x4*)(ga + 8);
      u32x4 b0 = *(const u32x4*)gb;
      u32x4 b1 = *(const u32x4*)(gb + 8);
      *(u32x4*)&As[lrow * GLDT + lseg]     = a0;
      *(u32x4*)&As[lrow * GLDT + lseg + 8] = a1;
      *(u32x4*)&Bs[lrow * GLDT + lseg]     = b0;
      *(u32x4*)&Bs[lrow * GLDT + lseg + 8] = b1;
    }
    __syncthreads();

    // A fragment (16x32): lanes 0-15 hold K {0..7,16..23}, lanes 16-31 {8..15,24..31}
    H16 af;
    {
      const half_t* p = &As[(wave * 16 + fr) * GLDT + fhi * 8];
      af.u[0] = *(const u32x4*)p;
      af.u[1] = *(const u32x4*)(p + 16);
    }
#pragma unroll
    for (int nt = 0; nt < 4; ++nt) {
      // B fragment (32x16): lanes 0-15 hold K {0..7,16..23}, lanes 16-31 {8..15,24..31}
      H16 bf;
      const half_t* p = &Bs[(nt * 16 + fr) * GLDT + fhi * 8];
      bf.u[0] = *(const u32x4*)p;
      bf.u[1] = *(const u32x4*)(p + 16);
      acc[nt] = __builtin_amdgcn_wmma_f32_16x16x32_f16(
          false, af.h, false, bf.h, (short)0, acc[nt], false, false);
    }
    __syncthreads();
  }

  const int rowAdd = (lane >> 4) * 8;
  const int colIn  = lane & 15;
#pragma unroll
  for (int nt = 0; nt < 4; ++nt) {
    int col = nBase + nt * 16 + colIn;
    float sc = scale ? scale[col] : 1.0f;
    float bs = bias  ? bias[col]  : 0.0f;
#pragma unroll
    for (int r = 0; r < 8; ++r) {
      int row = mBase + wave * 16 + rowAdd + r;
      float v = acc[nt][r] * sc + bs;
      size_t o = (size_t)row * N + col;
      if (outF) outF[o] = v;
      if (outH) outH[o] = (half_t)v;
    }
  }
}

// ---------------------------------------------------------------------
// Big-M WMMA GEMM: BM=BN=128, BK=32, 256 threads = 8 waves.
// Each wave owns a 32x64 sub-tile: 2 A-frags x 4 B-frags -> 8 WMMAs per
// 12 ds_load_b128 (vs 4 per 10 for the 64x64 tile): better LDS/WMMA ratio
// and half the barriers per MAC. Used for all 65536-row GEMMs.
__global__ __launch_bounds__(256)
void vltf_gemm_wmma128(const half_t* __restrict__ A, const half_t* __restrict__ Wt,
                       const float* __restrict__ bias, const float* __restrict__ scale,
                       float* __restrict__ outF, half_t* __restrict__ outH,
                       int M, int N, int K) {
  __shared__ alignas(16) half_t As[128 * GLDT];
  __shared__ alignas(16) half_t Bs[128 * GLDT];

  const int tid  = threadIdx.x;
  const int wave = tid >> 5;
  const int lane = tid & 31;
  const int wr = wave >> 1;           // 0..3 : 32-row group
  const int wc = wave & 1;            // 0..1 : 64-col group
  const int mBase = blockIdx.x * 128;
  const int nBase = blockIdx.y * 128;

  v8f acc[2][4];
#pragma unroll
  for (int i = 0; i < 2; ++i)
#pragma unroll
    for (int j = 0; j < 4; ++j)
#pragma unroll
      for (int e = 0; e < 8; ++e) acc[i][j][e] = 0.0f;

  const int lrow = tid >> 1;          // 0..127
  const int lseg = (tid & 1) * 16;    // 0 or 16 (halves)
  const int fr  = lane & 15;
  const int fhi = lane >> 4;

  for (int k0 = 0; k0 < K; k0 += 32) {
    // stage A tile (128x32) and W tile (128x32); 2 b128 loads each per thread
    {
      const half_t* ga = A  + (size_t)(mBase + lrow) * K + k0 + lseg;
      const half_t* gb = Wt + (size_t)(nBase + lrow) * K + k0 + lseg;
      u32x4 a0 = *(const u32x4*)ga;
      u32x4 a1 = *(const u32x4*)(ga + 8);
      u32x4 b0 = *(const u32x4*)gb;
      u32x4 b1 = *(const u32x4*)(gb + 8);
      *(u32x4*)&As[lrow * GLDT + lseg]     = a0;
      *(u32x4*)&As[lrow * GLDT + lseg + 8] = a1;
      *(u32x4*)&Bs[lrow * GLDT + lseg]     = b0;
      *(u32x4*)&Bs[lrow * GLDT + lseg + 8] = b1;
    }
    __syncthreads();

    H16 af[2];
#pragma unroll
    for (int mt = 0; mt < 2; ++mt) {
      const half_t* p = &As[(wr * 32 + mt * 16 + fr) * GLDT + fhi * 8];
      af[mt].u[0] = *(const u32x4*)p;
      af[mt].u[1] = *(const u32x4*)(p + 16);
    }
#pragma unroll
    for (int nt = 0; nt < 4; ++nt) {
      H16 bf;
      const half_t* p = &Bs[(wc * 64 + nt * 16 + fr) * GLDT + fhi * 8];
      bf.u[0] = *(const u32x4*)p;
      bf.u[1] = *(const u32x4*)(p + 16);
      acc[0][nt] = __builtin_amdgcn_wmma_f32_16x16x32_f16(
          false, af[0].h, false, bf.h, (short)0, acc[0][nt], false, false);
      acc[1][nt] = __builtin_amdgcn_wmma_f32_16x16x32_f16(
          false, af[1].h, false, bf.h, (short)0, acc[1][nt], false, false);
    }
    __syncthreads();
  }

  const int rowAdd = (lane >> 4) * 8;
  const int colIn  = lane & 15;
#pragma unroll
  for (int nt = 0; nt < 4; ++nt) {
    int col = nBase + wc * 64 + nt * 16 + colIn;
    float sc = scale ? scale[col] : 1.0f;
    float bs = bias  ? bias[col]  : 0.0f;
#pragma unroll
    for (int mt = 0; mt < 2; ++mt) {
#pragma unroll
      for (int r = 0; r < 8; ++r) {
        int row = mBase + wr * 32 + mt * 16 + rowAdd + r;
        float v = acc[mt][nt][r] * sc + bs;
        size_t o = (size_t)row * N + col;
        if (outF) outF[o] = v;
        if (outH) outH[o] = (half_t)v;
      }
    }
  }
}

// ---------------------------------------------------------------------
// q_mem = mem_tok @ mf_qw.T + mf_qb   (1x512, K=512)
__global__ __launch_bounds__(256)
void vltf_qmem_kernel(const float* __restrict__ mem, const float* __restrict__ w,
                      const float* __restrict__ bias, float* __restrict__ out) {
  int n = blockIdx.x * blockDim.x + threadIdx.x;
  if (n >= 512) return;
  const float* wr = w + (size_t)n * 512;
  float acc = bias[n];
  for (int c = 0; c < 512; ++c) acc += mem[c] * wr[c];
  out[n] = acc;
}

// ---------------------------------------------------------------------
// vl_att: softmax_j( (q*qmask) . k / 8 ), j over 4096 keys.
// One block per (b,h,qi); 256 threads, 16 keys each.
__global__ __launch_bounds__(256)
void vltf_vlatt_kernel(const float* __restrict__ q_l, const half_t* __restrict__ k_s,
                       const float* __restrict__ qmask, float* __restrict__ att_out) {
  int blk = blockIdx.x;
  int qi = blk % 20, h = (blk / 20) % 8, b = blk / 160;
  __shared__ float qv[64];
  __shared__ float red[256];
  int t = threadIdx.x;
  if (t < 64) qv[t] = q_l[(size_t)(b * 20 + qi) * 512 + h * 64 + t] * qmask[b * 20 + qi];
  __syncthreads();

  float logit[16];
#pragma unroll
  for (int n = 0; n < 16; ++n) {
    int j = t + n * 256;
    const half_t* kp = k_s + (size_t)(b * 4096 + j) * 512 + h * 64;
    float acc = 0.f;
    for (int d = 0; d < 64; ++d) acc += qv[d] * (float)kp[d];
    logit[n] = acc * 0.125f;
  }
  float m = logit[0];
#pragma unroll
  for (int n = 1; n < 16; ++n) m = fmaxf(m, logit[n]);
  red[t] = m; __syncthreads();
  for (int s = 128; s > 0; s >>= 1) { if (t < s) red[t] = fmaxf(red[t], red[t + s]); __syncthreads(); }
  m = red[0]; __syncthreads();
  float se = 0.f;
#pragma unroll
  for (int n = 0; n < 16; ++n) { logit[n] = expf(logit[n] - m); se += logit[n]; }
  red[t] = se; __syncthreads();
  for (int s = 128; s > 0; s >>= 1) { if (t < s) red[t] += red[t + s]; __syncthreads(); }
  float inv = 1.f / red[0];
  float* op = att_out + ((size_t)(b * 8 + h) * 20 + qi) * 4096;
#pragma unroll
  for (int n = 0; n < 16; ++n) op[t + n * 256] = logit[n] * inv;
}

// ---------------------------------------------------------------------
// Memory fuse: one block per batch, 512 threads.
__global__ __launch_bounds__(512)
void vltf_memfuse_kernel(const float* __restrict__ q_mem, const float* __restrict__ k_l,
                         const float* __restrict__ v_l, const float* __restrict__ kmask,
                         const float* __restrict__ ow, const float* __restrict__ ob,
                         const float* __restrict__ neg_tok,
                         float* __restrict__ mem_att_out, float* __restrict__ lan_mem_out,
                         float* __restrict__ lan_mem_ws) {
  int b = blockIdx.x, t = threadIdx.x;
  __shared__ float lg[8][20];
  __shared__ float att[8][20];
  __shared__ float ovec[512];
  if (t < 160) {
    int h = t / 20, j = t % 20;
    const float* kp = k_l + (size_t)(b * 20 + j) * 512 + h * 64;
    const float* qp = q_mem + h * 64;
    float acc = 0.f;
    for (int d = 0; d < 64; ++d) acc += qp[d] * kp[d];
    lg[h][j] = acc * kmask[b * 20 + j] * 0.125f;   // mask multiplies k -> scales logit
  }
  __syncthreads();
  if (t < 8) {
    int h = t;
    float m = lg[h][0];
    for (int j = 1; j < 20; ++j) m = fmaxf(m, lg[h][j]);
    float s = 0.f;
    for (int j = 0; j < 20; ++j) { float e = expf(lg[h][j] - m); att[h][j] = e; s += e; }
    float inv = 1.f / s;
    for (int j = 0; j < 20; ++j) {
      att[h][j] *= inv;
      mem_att_out[(size_t)(b * 8 + h) * 20 + j] = att[h][j];
    }
  }
  __syncthreads();
  {
    int h = t >> 6;
    float acc = 0.f;
    for (int j = 0; j < 20; ++j)
      acc += att[h][j] * v_l[(size_t)(b * 20 + j) * 512 + t] * kmask[b * 20 + j];
    ovec[t] = acc;
  }
  __syncthreads();
  {
    const float* wr = ow + (size_t)t * 512;
    float acc = ob[t];
    for (int c = 0; c < 512; ++c) acc += ovec[c] * wr[c];
    // lan_mem = inorm(concat([neg_tok, proj]), axis=tokens(2))
    float x0 = neg_tok[t], x1 = acc;
    float m  = 0.5f * (x0 + x1);
    float d0 = x0 - m, d1 = x1 - m;
    float var = 0.5f * (d0 * d0 + d1 * d1);
    float inv = rsqrtf(var + EPSN);
    float o0 = d0 * inv, o1 = d1 * inv;
    lan_mem_out[(size_t)(b * 2) * 512 + t]     = o0;
    lan_mem_out[(size_t)(b * 2 + 1) * 512 + t] = o1;
    lan_mem_ws[(size_t)(b * 2) * 512 + t]      = o0;
    lan_mem_ws[(size_t)(b * 2 + 1) * 512 + t]  = o1;
  }
}

// ---------------------------------------------------------------------
// k_m / v_m = lan_mem @ {ff_kw,ff_vw}.T + b    (32x512 each)
__global__ __launch_bounds__(256)
void vltf_kv_kernel(const float* __restrict__ lm,
                    const float* __restrict__ wk, const float* __restrict__ bk,
                    const float* __restrict__ wv, const float* __restrict__ bv,
                    float* __restrict__ k_m, float* __restrict__ v_m) {
  int idx = blockIdx.x * blockDim.x + threadIdx.x;
  if (idx >= 32768) return;
  int which = idx >> 14;            // 0 = k, 1 = v
  int id = idx & 16383;
  int n = id & 511, row = id >> 9;  // row = b*2 + token
  const float* w  = which ? wv : wk;
  const float* bb = which ? bv : bk;
  const float* x  = lm + (size_t)row * 512;
  const float* wr = w + (size_t)n * 512;
  float acc = bb[n];
  for (int c = 0; c < 512; ++c) acc += x[c] * wr[c];
  (which ? v_m : k_m)[(size_t)row * 512 + n] = acc;
}

// ---------------------------------------------------------------------
// Feature-fuse attention: 2 keys per vision token; writes f_att + attnout(f16).
// 256 threads = 32 tokens x 8 heads.
__global__ __launch_bounds__(256)
void vltf_ffatt_kernel(const half_t* __restrict__ q_s, const float* __restrict__ k_m,
                       const float* __restrict__ v_m, float* __restrict__ fatt,
                       half_t* __restrict__ attn) {
  int t = threadIdx.x;
  int h = t & 7, tl = t >> 3;
  size_t i = (size_t)blockIdx.x * 32 + tl;
  int b = (int)(i >> 12);
  const half_t* qp  = q_s + i * 512 + h * 64;
  const float*  k0p = k_m + (size_t)(b * 2) * 512 + h * 64;
  const float*  k1p = k0p + 512;
  float l0 = 0.f, l1 = 0.f;
  for (int d = 0; d < 64; ++d) { float q = (float)qp[d]; l0 += q * k0p[d]; l1 += q * k1p[d]; }
  l0 *= 0.125f; l1 *= 0.125f;
  float m = fmaxf(l0, l1);
  float e0 = expf(l0 - m), e1 = expf(l1 - m);
  float inv = 1.f / (e0 + e1);
  float a0 = e0 * inv, a1 = e1 * inv;
  size_t fo = ((size_t)(b * 8 + h) * 4096 + (i & 4095)) * 2;
  fatt[fo] = a0; fatt[fo + 1] = a1;
  const float* v0p = v_m + (size_t)(b * 2) * 512 + h * 64;
  const float* v1p = v0p + 512;
  half_t* op = attn + i * 512 + h * 64;
  for (int d = 0; d < 64; ++d) op[d] = (half_t)(a0 * v0p[d] + a1 * v1p[d]);
}

// ---------------------------------------------------------------------
// per-(b,c) mean / rstd over the 4096 token axis of s2 (f16)
__global__ __launch_bounds__(256)
void vltf_stats_kernel(const half_t* __restrict__ s2, float* __restrict__ mean,
                       float* __restrict__ rstd) {
  int b  = blockIdx.x >> 3;
  int cb = (blockIdx.x & 7) * 64;
  int t  = threadIdx.x;
  int c  = cb + (t & 63);
  int r0 = t >> 6;                       // 0..3
  float s = 0.f, q = 0.f;
  const half_t* base = s2 + (size_t)b * 4096 * 512 + c;
  for (int r = r0; r < 4096; r += 4) {
    float v = (float)base[(size_t)r * 512];
    s += v; q += v * v;
  }
  __shared__ float ss[256], sq[256];
  ss[t] = s; sq[t] = q; __syncthreads();
  if (t < 64) {
    float S = ss[t] + ss[t + 64] + ss[t + 128] + ss[t + 192];
    float Q = sq[t] + sq[t + 64] + sq[t + 128] + sq[t + 192];
    float mu = S * (1.f / 4096.f);
    float var = Q * (1.f / 4096.f) - mu * mu;
    mean[b * 512 + cb + t] = mu;
    rstd[b * 512 + cb + t] = rsqrtf(var + EPSN);
  }
}

// ---------------------------------------------------------------------
// s2n = (s2 - mean) * rstd   (f16 out)
__global__ __launch_bounds__(256)
void vltf_inorm_kernel(const half_t* __restrict__ s2, const float* __restrict__ mean,
                       const float* __restrict__ rstd, half_t* __restrict__ out, size_t n) {
  size_t idx = (size_t)blockIdx.x * blockDim.x + threadIdx.x;
  if (idx >= n) return;
  size_t i = idx >> 9;
  int c = (int)(idx & 511);
  int b = (int)(i >> 12);
  float v = ((float)s2[idx] - mean[b * 512 + c]) * rstd[b * 512 + c];
  out[idx] = (half_t)v;
}

// =====================================================================
extern "C" void kernel_launch(void* const* d_in, const int* in_sizes, int n_in,
                              void* d_out, int out_size, void* d_ws, size_t ws_size,
                              hipStream_t stream) {
  (void)in_sizes; (void)n_in; (void)out_size; (void)ws_size;
  const float* src      = (const float*)d_in[0];
  const float* lan      = (const float*)d_in[1];
  const float* lanmask  = (const float*)d_in[3];
  const float* mem_tok  = (const float*)d_in[4];
  const float* neg_tok  = (const float*)d_in[5];
  const float* in_w     = (const float*)d_in[6];
  const float* in_b     = (const float*)d_in[7];
  const float* in_bn_g  = (const float*)d_in[8];
  const float* in_bn_b  = (const float*)d_in[9];
  const float* lan_w    = (const float*)d_in[10];
  const float* lan_b    = (const float*)d_in[11];
  const float* out_w    = (const float*)d_in[12];
  const float* out_b    = (const float*)d_in[13];
  const float* out_bn_g = (const float*)d_in[14];
  const float* out_bn_b = (const float*)d_in[15];
  const float* vl_qw = (const float*)d_in[16]; const float* vl_qb = (const float*)d_in[17];
  const float* vl_kw = (const float*)d_in[18]; const float* vl_kb = (const float*)d_in[19];
  const float* mf_qw = (const float*)d_in[24]; const float* mf_qb = (const float*)d_in[25];
  const float* mf_kw = (const float*)d_in[26]; const float* mf_kb = (const float*)d_in[27];
  const float* mf_vw = (const float*)d_in[28]; const float* mf_vb = (const float*)d_in[29];
  const float* mf_ow = (const float*)d_in[30]; const float* mf_ob = (const float*)d_in[31];
  const float* ff_qw = (const float*)d_in[32]; const float* ff_qb = (const float*)d_in[33];
  const float* ff_kw = (const float*)d_in[34]; const float* ff_kb = (const float*)d_in[35];
  const float* ff_vw = (const float*)d_in[36]; const float* ff_vb = (const float*)d_in[37];
  const float* ff_ow = (const float*)d_in[38]; const float* ff_ob = (const float*)d_in[39];

  // output regions
  float* out0   = (float*)d_out;
  float* out_lm = out0 + 33554432ull;
  float* out_vl = out_lm + 16384ull;
  float* out_ma = out_vl + 10485760ull;
  float* out_fa = out_ma + 2560ull;

  // workspace bump allocator (256B aligned)
  char* wsb = (char*)d_ws;
  size_t off = 0;
  auto alloc = [&](size_t bytes) -> void* {
    void* p = wsb + off;
    off = (off + bytes + 255) & ~(size_t)255;
    return p;
  };

  const size_t MT = (size_t)65536;                 // B*HW
  const size_t BIGB = MT * 512 * sizeof(half_t);   // 64 MB
  half_t* R1 = (half_t*)alloc(BIGB);
  half_t* R2 = (half_t*)alloc(BIGB);
  half_t* R3 = (half_t*)alloc(BIGB);
  // region reuse schedule (lifetimes verified against launch order below):
  half_t* src_h = R1;  half_t* s_h  = R2;  half_t* ks_h  = R3;
  half_t* qs_h  = R3;  half_t* attn_h = R1; half_t* s2_h = R2;  half_t* s2n_h = R3;

  half_t* lan_h   = (half_t*)alloc(245760 * sizeof(half_t));
  half_t* l_h     = (half_t*)alloc(320 * 512 * sizeof(half_t));
  half_t* in_w_h  = (half_t*)alloc(262144 * sizeof(half_t));
  half_t* lan_w_h = (half_t*)alloc(393216 * sizeof(half_t));
  half_t* vl_qw_h = (half_t*)alloc(262144 * sizeof(half_t));
  half_t* vl_kw_h = (half_t*)alloc(262144 * sizeof(half_t));
  half_t* mf_kw_h = (half_t*)alloc(262144 * sizeof(half_t));
  half_t* mf_vw_h = (half_t*)alloc(262144 * sizeof(half_t));
  half_t* ff_qw_h = (half_t*)alloc(262144 * sizeof(half_t));
  half_t* ff_ow_h = (half_t*)alloc(262144 * sizeof(half_t));
  half_t* out_w_h = (half_t*)alloc(262144 * sizeof(half_t));
  float* q_l   = (float*)alloc(320 * 512 * sizeof(float));
  float* k_l   = (float*)alloc(320 * 512 * sizeof(float));
  float* v_l   = (float*)alloc(320 * 512 * sizeof(float));
  float* q_mem = (float*)alloc(512 * sizeof(float));
  float* lm_ws = (float*)alloc(16 * 2 * 512 * sizeof(float));
  float* k_m   = (float*)alloc(16 * 2 * 512 * sizeof(float));
  float* v_m   = (float*)alloc(16 * 2 * 512 * sizeof(float));
  float* meanv = (float*)alloc(16 * 512 * sizeof(float));
  float* rstdv = (float*)alloc(16 * 512 * sizeof(float));
  float* alpha1 = (float*)alloc(512 * sizeof(float));
  float* beta1  = (float*)alloc(512 * sizeof(float));
  float* alpha2 = (float*)alloc(512 * sizeof(float));
  float* beta2  = (float*)alloc(512 * sizeof(float));

  auto cvt = [&](const float* s, half_t* d, size_t n) {
    vltf_cvt_kernel<<<(unsigned)((n + 255) / 256), 256, 0, stream>>>(s, d, n);
  };
  auto gemmS = [&](const half_t* A, const half_t* Wt, const float* bias, const float* scale,
                   float* oF, half_t* oH, int M, int N, int K) {
    dim3 g(M / 64, N / 64);
    vltf_gemm_wmma64<<<g, 128, 0, stream>>>(A, Wt, bias, scale, oF, oH, M, N, K);
  };
  auto gemmB = [&](const half_t* A, const half_t* Wt, const float* bias, const float* scale,
                   float* oF, half_t* oH, int M, int N, int K) {
    dim3 g(M / 128, N / 128);
    vltf_gemm_wmma128<<<g, 256, 0, stream>>>(A, Wt, bias, scale, oF, oH, M, N, K);
  };

  // --- BN folding constants
  vltf_prep_kernel<<<2, 256, 0, stream>>>(in_bn_g, in_b, in_bn_b,
                                          out_bn_g, out_b, out_bn_b,
                                          alpha1, beta1, alpha2, beta2);
  // --- f32 -> f16 casts (activations + GEMM weights)
  cvt(src,   src_h,   (size_t)33554432);
  cvt(lan,   lan_h,   (size_t)245760);
  cvt(in_w,  in_w_h,  (size_t)262144);
  cvt(lan_w, lan_w_h, (size_t)393216);
  cvt(vl_qw, vl_qw_h, (size_t)262144);
  cvt(vl_kw, vl_kw_h, (size_t)262144);
  cvt(mf_kw, mf_kw_h, (size_t)262144);
  cvt(mf_vw, mf_vw_h, (size_t)262144);
  cvt(ff_qw, ff_qw_h, (size_t)262144);
  cvt(ff_ow, ff_ow_h, (size_t)262144);
  cvt(out_w, out_w_h, (size_t)262144);

  // --- language path: l = lan @ lan_w.T + lan_b   (320x512, K=768)
  gemmS(lan_h, lan_w_h, lan_b, nullptr, nullptr, l_h, 320, 512, 768);
  // --- s = BN(src @ in_w.T + in_b)   (65536x512) -> f16
  gemmB(src_h, in_w_h, beta1, alpha1, nullptr, s_h, 65536, 512, 512);
  // --- small projections from l
  gemmS(l_h, vl_qw_h, vl_qb, nullptr, q_l, nullptr, 320, 512, 512);
  gemmS(l_h, mf_kw_h, mf_kb, nullptr, k_l, nullptr, 320, 512, 512);
  gemmS(l_h, mf_vw_h, mf_vb, nullptr, v_l, nullptr, 320, 512, 512);
  // --- vl keys: k_s = s @ vl_kw.T + vl_kb -> f16 (R3)
  gemmB(s_h, vl_kw_h, vl_kb, nullptr, nullptr, ks_h, 65536, 512, 512);
  // --- vl_att output (k_s dead afterwards -> R3 reusable)
  vltf_vlatt_kernel<<<2560, 256, 0, stream>>>(q_l, ks_h, lanmask, out_vl);
  // --- memory fuse
  vltf_qmem_kernel<<<2, 256, 0, stream>>>(mem_tok, mf_qw, mf_qb, q_mem);
  vltf_memfuse_kernel<<<16, 512, 0, stream>>>(q_mem, k_l, v_l, lanmask,
                                              mf_ow, mf_ob, neg_tok,
                                              out_ma, out_lm, lm_ws);
  vltf_kv_kernel<<<128, 256, 0, stream>>>(lm_ws, ff_kw, ff_kb, ff_vw, ff_vb, k_m, v_m);
  // --- feature fuse: q_s = s @ ff_qw.T + ff_qb -> f16 (R3; s dead afterwards)
  gemmB(s_h, ff_qw_h, ff_qb, nullptr, nullptr, qs_h, 65536, 512, 512);
  vltf_ffatt_kernel<<<2048, 256, 0, stream>>>(qs_h, k_m, v_m, out_fa, attn_h);
  // --- s2 = attn @ ff_ow.T + ff_ob -> f16 (R2)
  gemmB(attn_h, ff_ow_h, ff_ob, nullptr, nullptr, s2_h, 65536, 512, 512);
  // --- instance norm over the 4096-token axis
  vltf_stats_kernel<<<128, 256, 0, stream>>>(s2_h, meanv, rstdv);
  vltf_inorm_kernel<<<131072, 256, 0, stream>>>(s2_h, meanv, rstdv, s2n_h,
                                                (size_t)33554432);
  // --- out = BN(s2n @ out_w.T + out_b) -> f32 d_out
  gemmB(s2n_h, out_w_h, beta2, alpha2, out0, nullptr, 65536, 512, 512);
}
